// CustomizedConvolutionModel_13975823581603
// MI455X (gfx1250) — matrix operand order repaired and checked by reading, not compiled, not run on hardware
//
#include <hip/hip_runtime.h>
#include <hip/hip_bf16.h>

// ---------------------------------------------------------------------------
// CDNA5 (gfx1250) forward pass for the reference CNN.
// Matrix work: v_wmma_f32_16x16x32_bf16 (bf16 operands, f32 accumulate).
// ---------------------------------------------------------------------------

typedef __attribute__((ext_vector_type(16))) __bf16 v16bf;
typedef __attribute__((ext_vector_type(8)))  float  v8f;
typedef __attribute__((ext_vector_type(4)))  int    v4i;

// gfx1250 async global->LDS path (probe-confirmed: params are generic int*).
#if defined(__AMDGCN__) && __has_builtin(__builtin_amdgcn_global_load_async_to_lds_b32)
#define HAVE_ASYNC_LDS 1
#else
#define HAVE_ASYNC_LDS 0
#endif

__device__ __forceinline__ void wait_asynccnt0() {
#if __has_builtin(__builtin_amdgcn_s_wait_asynccnt)
  __builtin_amdgcn_s_wait_asynccnt(0);
#else
  asm volatile("s_wait_asynccnt 0x0" ::: "memory");
#endif
}

// A-fragment (16x32, 16-bit) element->k map used in this file:
//   k(e) = (lane&16) + e, row = lane&15 (same order as B).
// B-fragment (32x16), same order as A here: k(e) = (lane&16) + e, column = lane&15.
// C/D: column = lane&15, row = e + 8*(lane>>4).

__device__ __forceinline__ v8f wmma_bf16(v16bf a, v16bf b, v8f c) {
  return __builtin_amdgcn_wmma_f32_16x16x32_bf16(
      /*neg_a=*/false, a, /*neg_b=*/false, b,
      /*c_mod=*/(short)0, c, /*reuse_a=*/false, /*reuse_b=*/false);
}

// ---------------------------------------------------------------------------
// Implicit-GEMM 3x3 SAME conv + bias + ReLU, NHWC, fp32 in/out, bf16 WMMA.
// Grid: x = ceil(Nimg*H*W / 64), y = ceil(Cout/16). Block: 128 (4 waves).
// K = 9*Cin (max 729 -> Kp max 736).
//  - Weights async-staged to LDS, converted to bf16 n-major so each lane's
//    B fragment is one 32B vector LDS read.
//  - Per-k im2col decode precomputed into an LDS table {p=kh*3+kw, int16 off};
//    per-pixel 9-tap validity is a bitmask computed once -> inner-loop test
//    is shift+and, all gathers branchless.
//  - All addressing is i32; output address is m*Cout + cout (m = linear
//    NHWC pixel index). Epilogue has a wave-uniform full-tile fast path.
// ---------------------------------------------------------------------------
#define CONV_KP_MAX 736

__global__ __launch_bounds__(128)
void conv3x3_wmma(const float* __restrict__ X, int xBatchStride,
                  const float* __restrict__ Wt, const float* __restrict__ bias,
                  float* __restrict__ Y,
                  int Nimg, int H, int Wd, int Cin, int Cout) {
  __shared__ __attribute__((aligned(32))) float  sWf[CONV_KP_MAX * 16];
  __shared__ __attribute__((aligned(32))) __bf16 sWb[16 * CONV_KP_MAX];
  __shared__ __attribute__((aligned(32))) int    sTab[CONV_KP_MAX];

  const int K   = 9 * Cin;
  const int Kp  = (K + 31) & ~31;
  const int tid = threadIdx.x;
  const int co0 = blockIdx.y * 16;

  // (1) Stage one 16-wide Cout slice of the weights into LDS (raw 4B words).
  for (int idx = tid; idx < Kp * 16; idx += 128) {
    int k = idx >> 4, co = idx & 15;
    bool vok = (k < K) && (co0 + co) < Cout;
#if HAVE_ASYNC_LDS
    if (vok)
      __builtin_amdgcn_global_load_async_to_lds_b32(
          (int*)(Wt + k * Cout + co0 + co), (int*)&sWf[idx], 0, 0);
#else
    sWf[idx] = vok ? Wt[k * Cout + co0 + co] : 0.0f;
#endif
  }
  // (2) im2col decode table: bits[19:16] = p (kh*3+kw; 15 = invalid pad),
  //     bits[15:0] = int16 relative offset.
  for (int k = tid; k < Kp; k += 128) {
    int v;
    if (k < K) {
      int p  = k / Cin, ci = k - p * Cin;
      int kh = p / 3,   kw = p - kh * 3;
      int off = ((kh - 1) * Wd + (kw - 1)) * Cin + ci;  // |off| < 32768
      v = (p << 16) | (off & 0xFFFF);
    } else {
      v = 15 << 16;
    }
    sTab[k] = v;
  }
#if HAVE_ASYNC_LDS
  wait_asynccnt0();
#endif
  __syncthreads();
  // (3) Convert to bf16, n-major; zero the padded/invalid entries.
  for (int idx = tid; idx < Kp * 16; idx += 128) {
    int k = idx >> 4, co = idx & 15;
    bool vok = (k < K) && (co0 + co) < Cout;
    float v = vok ? sWf[idx] : 0.0f;
    sWb[co * CONV_KP_MAX + k] = (__bf16)v;
  }
  __syncthreads();

  const int lane = tid & 31;
  const int wv   = tid >> 5;
  const int HW   = H * Wd;
  const int M    = Nimg * HW;
  const int m0   = blockIdx.x * 64 + wv * 16;
  const int mrow = m0 + (lane & 15);
  const bool mok = (mrow < M);

  // Decode (img, h, w) once per lane, u32 math.
  int mr = mok ? mrow : (M - 1);
  int img = (int)((unsigned)mr / (unsigned)HW);
  int rr  = mr - img * HW;
  int h   = (int)((unsigned)rr / (unsigned)Wd);
  int w   = rr - h * Wd;

  // 9-tap validity bitmask for this pixel (bit p = kh*3+kw).
  unsigned vmask = 0;
  if (mok) {
#pragma unroll
    for (int kh = 0; kh < 3; ++kh)
#pragma unroll
      for (int kw = 0; kw < 3; ++kw) {
        int hh = h + kh - 1, ww = w + kw - 1;
        if (hh >= 0 && hh < H && ww >= 0 && ww < Wd)
          vmask |= 1u << (kh * 3 + kw);
      }
  }

  const float* Xb = X + img * xBatchStride;
  const int pixBase = (h * Wd + w) * Cin;
  const __bf16* bcol = &sWb[(lane & 15) * CONV_KP_MAX];
  const int bSel = lane & 16;
  const int hi8  = (lane & 16) >> 1;

  v8f acc = {0.f, 0.f, 0.f, 0.f, 0.f, 0.f, 0.f, 0.f};

  for (int k0 = 0; k0 < Kp; k0 += 32) {
    // B fragment: 16 consecutive bf16 (32B, 32B-aligned) from LDS.
    v16bf b = *(const v16bf*)(bcol + k0 + bSel);
    // Table entries for this lane: one contiguous 16-int run (4x 16B loads).
    v4i t0 = *(const v4i*)&sTab[k0 + bSel];
    v4i t1 = *(const v4i*)&sTab[k0 + bSel + 4];
    v4i t2 = *(const v4i*)&sTab[k0 + bSel + 8];
    v4i t3 = *(const v4i*)&sTab[k0 + bSel + 12];
    int tt[16];
#pragma unroll
    for (int q = 0; q < 4; ++q) {
      tt[q] = t0[q]; tt[4 + q] = t1[q]; tt[8 + q] = t2[q]; tt[12 + q] = t3[q];
    }
    v16bf a;
#pragma unroll
    for (int e = 0; e < 16; ++e) {
      int t = tt[e];
      bool ok = (vmask >> ((t >> 16) & 15)) & 1;   // p=15 (pad) -> 0
      int ixo = ok ? (pixBase + (int)(short)(t & 0xFFFF)) : 0;
      float v = Xb[ixo];
      a[e] = (__bf16)(ok ? v : 0.0f);
    }
    acc = wmma_bf16(a, b, acc);
  }

  // Epilogue: D column = lane&15, row = e + 8*(lane>>4); Y addr = m*Cout+cout.
  const int cout = co0 + (lane & 15);
  const int hi   = lane >> 4;
  if (cout < Cout) {
    const float bv = bias[cout];                // load bias once
    if (m0 + 16 <= M) {
      // Full tile (wave-uniform): 8 unguarded stores.
      float* Yp = Y + (m0 + 8 * hi) * Cout + cout;
#pragma unroll
      for (int e = 0; e < 8; ++e)
        Yp[e * Cout] = fmaxf(acc[e] + bv, 0.0f);
    } else {
#pragma unroll
      for (int e = 0; e < 8; ++e) {
        int m = m0 + e + 8 * hi;
        if (m < M) Y[m * Cout + cout] = fmaxf(acc[e] + bv, 0.0f);
      }
    }
  }
}

// ---------------------------------------------------------------------------
// Dense GEMM, M = 16 (batch): Y[16,N] = relu(X[16,K] @ W[K,N] + b).
// Grid.x = ceil(N/16); block = 256 (8 waves split K, LDS reduce).
// Wave-uniform fast path has zero bounds selects: out-of-range columns are
// clamped to a safe address and simply never stored (D column n lives only
// in lane n, whose store is masked). All indices i32.
// ---------------------------------------------------------------------------
__global__ __launch_bounds__(256)
void gemm16_wmma(const float* __restrict__ X, int ldx,
                 const float* __restrict__ W, int ldw,
                 const float* __restrict__ bias,
                 float* __restrict__ Y, int ldy,
                 int K, int N, int relu) {
  __shared__ float red[8 * 32 * 8];

  const int tid  = threadIdx.x;
  const int lane = tid & 31;
  const int wv   = tid >> 5;
  const int n0   = blockIdx.x * 16;
  const int Kp   = (K + 31) & ~31;
  const int mrow = lane & 15;
  const int ncol = n0 + (lane & 15);
  const bool nok = (ncol < N);
  const int ncolL = nok ? ncol : (N - 1);
  const float* Wc = W + ncolL;
  const int hi8  = (lane & 16) >> 1;
  const int bSel = lane & 16;

  v8f acc = {0.f, 0.f, 0.f, 0.f, 0.f, 0.f, 0.f, 0.f};

  for (int k0 = wv * 32; k0 < Kp; k0 += 8 * 32) {
    // Prefetch next weight chunk for this wave (global_prefetch_b8).
    __builtin_prefetch(Wc + (k0 + 256) * ldw, 0, 1);
    const float* Xr = X + mrow * ldx + k0;
    v16bf a, b;
    if (k0 + 32 <= K) {          // wave-uniform fast path, no selects
#pragma unroll
      for (int e = 0; e < 16; ++e) {
        int kk = bSel + e;
        a[e] = (__bf16)Xr[kk];
        b[e] = (__bf16)Wc[(k0 + bSel + e) * ldw];
      }
    } else {                     // K tail: branchless selects
#pragma unroll
      for (int e = 0; e < 16; ++e) {
        int kk = bSel + e;
        bool okA = (k0 + kk) < K;
        float va = Xr[okA ? kk : -k0];
        a[e] = (__bf16)(okA ? va : 0.0f);

        int kb = k0 + bSel + e;
        bool okB = kb < K;
        float vb = Wc[okB ? (kb * ldw) : 0];
        b[e] = (__bf16)(okB ? vb : 0.0f);
      }
    }
    acc = wmma_bf16(a, b, acc);
  }

#pragma unroll
  for (int e = 0; e < 8; ++e) red[(wv * 32 + lane) * 8 + e] = acc[e];
  __syncthreads();

  if (wv == 0 && nok) {
    const int hi = lane >> 4;
    const float bv = bias[ncol];               // load bias once
    float* Yp = Y + 8 * hi * ldy + ncol;
#pragma unroll
    for (int e = 0; e < 8; ++e) {
      float s = 0.0f;
#pragma unroll
      for (int w2 = 0; w2 < 8; ++w2) s += red[(w2 * 32 + lane) * 8 + e];
      float v = s + bv;
      if (relu) v = fmaxf(v, 0.0f);
      Yp[e * ldy] = v;
    }
  }
}

// ---------------------------------------------------------------------------
// Elementwise / data-movement kernels.
// ---------------------------------------------------------------------------

// cropped[n,r,j,c] = images[n,2,42+r,9+j,c]; images (16,3,400,120,3).
__global__ void extract_crop(const float* __restrict__ img, float* __restrict__ out) {
  long i = (long)blockIdx.x * blockDim.x + threadIdx.x;
  const long total = 16L * 316 * 102 * 3;
  if (i >= total) return;
  int c = (int)(i % 3); long t = i / 3;
  int j = (int)(t % 102); t /= 102;
  int r = (int)(t % 316);
  int n = (int)(t / 316);
  out[i] = img[((long)(n * 3 + 2) * 400 + 42 + r) * 360 + (long)(9 + j) * 3 + c];
}

// Per-image min/max over 316*102*3 elements. Grid.x = 16, block = 256.
__global__ void pil_minmax(const float* __restrict__ X, float* __restrict__ mm) {
  __shared__ float smn[256], smx[256];
  const int n = blockIdx.x, tid = threadIdx.x;
  const long cnt = 316L * 102 * 3;
  const float* p = X + (long)n * cnt;
  float mn = 3.4e38f, mx = -3.4e38f;
  for (long i = tid; i < cnt; i += 256) {
    float v = p[i];
    mn = fminf(mn, v);
    mx = fmaxf(mx, v);
  }
  smn[tid] = mn; smx[tid] = mx;
  __syncthreads();
  for (int s = 128; s > 0; s >>= 1) {
    if (tid < s) {
      smn[tid] = fminf(smn[tid], smn[tid + s]);
      smx[tid] = fmaxf(smx[tid], smx[tid + s]);
    }
    __syncthreads();
  }
  if (tid == 0) { mm[2 * n] = smn[0]; mm[2 * n + 1] = smx[0]; }
}

// q = floor( where(max'!=0, x'/max', x') * 255 ), x' = x + max(-min,0).
__global__ void pil_quant(const float* __restrict__ X, const float* __restrict__ mm,
                          float* __restrict__ q) {
  long i = (long)blockIdx.x * blockDim.x + threadIdx.x;
  const long per = 316L * 102 * 3;
  const long total = 16 * per;
  if (i >= total) return;
  int n = (int)(i / per);
  float s = fmaxf(-mm[2 * n], 0.0f);
  float m = mm[2 * n + 1] + s;
  float xp = X[i] + s;
  float v = (m != 0.0f) ? (xp / m) * 255.0f : xp * 255.0f;
  q[i] = floorf(v);
}

// Depthwise 3x3 VALID conv on q, rounded/clipped, interior write; border = q.
__global__ void pil_dw(const float* __restrict__ q, float* __restrict__ out,
                       int which, float scale, float offset) {
  long i = (long)blockIdx.x * blockDim.x + threadIdx.x;
  const long total = 16L * 316 * 102 * 3;
  if (i >= total) return;
  int c = (int)(i % 3); long t = i / 3;
  int j = (int)(t % 102); t /= 102;
  int r = (int)(t % 316);
  int n = (int)(t / 316);
  const float* qb = q + (long)n * (316L * 102 * 3);
  float res;
  if (r >= 1 && r <= 314 && j >= 1 && j <= 100) {
    float center = (which == 0) ? 32.0f : 8.0f;
    float edge   = (which == 0) ? -2.0f : -1.0f;
    float acc = 0.0f;
#pragma unroll
    for (int dr = -1; dr <= 1; ++dr)
#pragma unroll
      for (int dc = -1; dc <= 1; ++dc) {
        float kk = (dr == 0 && dc == 0) ? center : edge;
        acc += qb[((long)(r + dr) * 102 + (j + dc)) * 3 + c] * kk;
      }
    float v = rintf(acc / scale + offset);   // round half to even (jnp.round)
    res = fminf(fmaxf(v, 0.0f), 255.0f);
  } else {
    res = qb[((long)r * 102 + j) * 3 + c];
  }
  out[i] = res;
}

// 2x2 max pool, stride 2, VALID, NHWC.
__global__ void maxpool2(const float* __restrict__ X, float* __restrict__ Y,
                         int Nimg, int H, int Wd, int C) {
  int Ho = H / 2, Wo = Wd / 2;
  long total = (long)Nimg * Ho * Wo * C;
  long i = (long)blockIdx.x * blockDim.x + threadIdx.x;
  if (i >= total) return;
  int c = (int)(i % C); long t = i / C;
  int wo = (int)(t % Wo); t /= Wo;
  int ho = (int)(t % Ho);
  int n  = (int)(t / Ho);
  const float* p = X + (((long)n * H + 2 * ho) * Wd + 2 * wo) * C + c;
  long rs = (long)Wd * C;
  float v = fmaxf(fmaxf(p[0], p[C]), fmaxf(p[rs], p[rs + C]));
  Y[i] = v;
}

// Segment gather: out (1824, 10, OW, 3); patch p = b*114 + cb*19 + rb;
// source pixel = src[b, 17*rb + ri, 17*cb + colStart + j, c]; j >= srcValid -> 0.
__global__ void seg_gather(const float* __restrict__ src, float* __restrict__ out,
                           int colStart, int OW, int srcValid) {
  long total = 1824L * 10 * OW * 3;
  long i = (long)blockIdx.x * blockDim.x + threadIdx.x;
  if (i >= total) return;
  int c = (int)(i % 3); long t = i / 3;
  int j  = (int)(t % OW); t /= OW;
  int ri = (int)(t % 10);
  int p  = (int)(t / 10);
  int b  = p / 114, pl = p - b * 114;
  int cb = pl / 19, rb = pl - cb * 19;
  float v = 0.0f;
  if (j < srcValid) {
    int row = 17 * rb + ri;
    int col = 17 * cb + colStart + j;
    v = src[(((long)b * 316 + row) * 102 + col) * 3 + c];
  }
  out[i] = v;
}

// Pack seg features: feat (16, 92340) from segPool (1824,5,9,9) and
// n{1,2,3}Pool (1824,5,3,9).
__global__ void seg_pack(const float* __restrict__ segPool,
                         const float* __restrict__ n1p,
                         const float* __restrict__ n2p,
                         const float* __restrict__ n3p,
                         float* __restrict__ feat) {
  long total = 16L * 92340;
  long i = (long)blockIdx.x * blockDim.x + threadIdx.x;
  if (i >= total) return;
  int ch  = (int)(i % 9); long t = i / 9;
  int col = (int)(t % 9); t /= 9;
  int row = (int)(t % 10); t /= 10;
  int pl  = (int)(t % 114);
  int b   = (int)(t / 114);
  long p = (long)b * 114 + pl;
  float v;
  if (row < 5) {
    v = segPool[((p * 5 + row) * 9 + col) * 9 + ch];
  } else {
    int r = row - 5;
    if (col < 3)      v = n1p[((p * 5 + r) * 3 + col) * 9 + ch];
    else if (col < 6) v = n2p[((p * 5 + r) * 3 + (col - 3)) * 9 + ch];
    else              v = n3p[((p * 5 + r) * 3 + (col - 6)) * 9 + ch];
  }
  feat[i] = v;
}

// ---------------------------------------------------------------------------
// Host orchestration.
// ---------------------------------------------------------------------------
extern "C" void kernel_launch(void* const* d_in, const int* in_sizes, int n_in,
                              void* d_out, int out_size, void* d_ws, size_t ws_size,
                              hipStream_t stream) {
  (void)in_sizes; (void)n_in; (void)out_size; (void)ws_size;

  // ---- Input unpack (JAX pytree: dict keys alphabetical) ----
  // d_in[0] = images; then ending (5x2), integ (8x4x2), overall (5x9x2),
  // seg (3x3x2). Tuples flatten as (W, b).
  const float* images = (const float*)d_in[0];
  int ix = 1;
  const float *endW[5], *endB[5];
  for (int i = 0; i < 5; ++i) { endW[i] = (const float*)d_in[ix++]; endB[i] = (const float*)d_in[ix++]; }
  const float *integW[8][4], *integB[8][4];
  for (int i = 0; i < 8; ++i)
    for (int j = 0; j < 4; ++j) { integW[i][j] = (const float*)d_in[ix++]; integB[i][j] = (const float*)d_in[ix++]; }
  const float *ovW[5][9], *ovB[5][9];
  for (int i = 0; i < 5; ++i)
    for (int j = 0; j < 9; ++j) { ovW[i][j] = (const float*)d_in[ix++]; ovB[i][j] = (const float*)d_in[ix++]; }
  const float *sgW[3][3], *sgB[3][3];
  for (int i = 0; i < 3; ++i)
    for (int j = 0; j < 3; ++j) { sgW[i][j] = (const float*)d_in[ix++]; sgB[i][j] = (const float*)d_in[ix++]; }

  // ---- Workspace bump allocator ----
  size_t off = 0;
  auto alloc = [&](size_t nfloats) -> float* {
    float* p = (float*)((char*)d_ws + off);
    off += ((nfloats * sizeof(float) + 255) & ~(size_t)255);
    return p;
  };
  const long CROPSZ = 16L * 316 * 102 * 3;     // 1,547,136
  float* cropped = alloc(CROPSZ);
  float* qbuf    = alloc(CROPSZ);
  float* sharpen = alloc(CROPSZ);
  float* contour = alloc(CROPSZ);
  float* mm      = alloc(32);
  float* bufA    = alloc(16L * 400 * 120 * 9); // 6,912,000 (max activation)
  float* bufB    = alloc(16L * 400 * 120 * 9);
  float* featO[5];
  featO[0] = alloc(16L * 60750);
  featO[1] = alloc(16L * 60750);
  featO[2] = alloc(16L * 37908);
  featO[3] = alloc(16L * 37908);
  featO[4] = alloc(16L * 37908);
  float* segIn   = alloc(1824L * 10 * 18 * 3);
  float* segPool = alloc(1824L * 5 * 9 * 9);
  float* nPool[3];
  for (int s = 0; s < 3; ++s) nPool[s] = alloc(1824L * 5 * 3 * 9);
  float* featS[3];
  for (int v = 0; v < 3; ++v) featS[v] = alloc(16L * 92340);
  float* t300a = alloc(16L * 300);
  float* t300b = alloc(16L * 300);
  float* concatBuf = alloc(16L * 2400);
  float* e200 = alloc(16L * 200);
  float* e100 = alloc(16L * 100);
  float* e50  = alloc(16L * 50);
  float* e20  = alloc(16L * 20);

  // ---- Launch helpers ----
  auto gridFor = [](long total) { return (unsigned)((total + 255) / 256); };
  auto conv = [&](const float* X, int bstr, const float* Wt, const float* B,
                  float* Y, int Nimg, int H, int Wd, int Cin, int Cout) {
    long M = (long)Nimg * H * Wd;
    dim3 grid((unsigned)((M + 63) / 64), (unsigned)((Cout + 15) / 16));
    conv3x3_wmma<<<grid, dim3(128), 0, stream>>>(X, bstr, Wt, B, Y, Nimg, H, Wd, Cin, Cout);
  };
  auto pool = [&](const float* X, float* Y, int Nimg, int H, int Wd, int C) {
    long total = (long)Nimg * (H / 2) * (Wd / 2) * C;
    maxpool2<<<gridFor(total), 256, 0, stream>>>(X, Y, Nimg, H, Wd, C);
  };
  auto gemm = [&](const float* X, int ldx, const float* Wt, const float* B,
                  float* Y, int ldy, int K, int N) {
    gemm16_wmma<<<dim3((unsigned)((N + 15) / 16)), 256, 0, stream>>>(
        X, ldx, Wt, N, B, Y, ldy, K, N, 1);
  };

  // ---- 1) Crop + PIL filters ----
  extract_crop<<<gridFor(CROPSZ), 256, 0, stream>>>(images, cropped);
  pil_minmax<<<16, 256, 0, stream>>>(cropped, mm);
  pil_quant<<<gridFor(CROPSZ), 256, 0, stream>>>(cropped, mm, qbuf);
  pil_dw<<<gridFor(CROPSZ), 256, 0, stream>>>(qbuf, sharpen, 0, 16.0f, 0.0f);
  pil_dw<<<gridFor(CROPSZ), 256, 0, stream>>>(qbuf, contour, 1, 1.0f, 255.0f);

  // ---- 2) Five "overall" conv blocks ----
  const int PLANE = 400 * 120 * 3;             // 144,000
  struct OvIn { const float* base; int bstr; int H, W; };
  OvIn ov[5] = {
      { images,            3 * PLANE, 400, 120 },   // original = images[:,0]
      { images + PLANE,    3 * PLANE, 400, 120 },   // color    = images[:,1]
      { cropped,           316 * 102 * 3, 316, 102 },
      { sharpen,           316 * 102 * 3, 316, 102 },
      { contour,           316 * 102 * 3, 316, 102 },
  };
  const int chIn [9] = { 3, 9, 9,  9, 27, 27, 27, 81, 81 };
  const int chOut[9] = { 9, 9, 9, 27, 27, 27, 81, 81, 81 };
  for (int b = 0; b < 5; ++b) {
    int H = ov[b].H, W = ov[b].W;
    const float* cur = ov[b].base;
    int curStride = ov[b].bstr;
    int li = 0;
    for (int stage = 0; stage < 3; ++stage) {
      conv(cur, curStride, ovW[b][li], ovB[b][li], bufA, 16, H, W, chIn[li], chOut[li]); ++li;
      conv(bufA, H * W * chIn[li], ovW[b][li], ovB[b][li], bufB, 16, H, W, chIn[li], chOut[li]); ++li;
      conv(bufB, H * W * chIn[li], ovW[b][li], ovB[b][li], bufA, 16, H, W, chIn[li], chOut[li]); ++li;
      int C = chOut[li - 1];
      float* dst = (stage == 2) ? featO[b] : bufB;
      pool(bufA, dst, 16, H, W, C);
      H /= 2; W /= 2;
      cur = dst; curStride = H * W * C;
    }
  }

  // ---- 3) Seg blocks (cropped, sharpen, contour) ----
  const float* segSrc[3] = { cropped, sharpen, contour };
  for (int v = 0; v < 3; ++v) {
    // Padded full-width branch: (1824, 10, 18, 3) -> pool (1824, 5, 9, 9)
    seg_gather<<<gridFor(1824L * 10 * 18 * 3), 256, 0, stream>>>(segSrc[v], segIn, 0, 18, 17);
    conv(segIn, 10 * 18 * 3, sgW[v][0], sgB[v][0], bufA, 1824, 10, 18, 3, 9);
    conv(bufA, 10 * 18 * 9, sgW[v][1], sgB[v][1], bufB, 1824, 10, 18, 9, 9);
    conv(bufB, 10 * 18 * 9, sgW[v][2], sgB[v][2], bufA, 1824, 10, 18, 9, 9);
    pool(bufA, segPool, 1824, 10, 18, 9);
    // Three narrow branches: cols 0:6, 5:11, 11:17 -> pool (1824, 5, 3, 9)
    const int cstart[3] = { 0, 5, 11 };
    for (int s = 0; s < 3; ++s) {
      seg_gather<<<gridFor(1824L * 10 * 6 * 3), 256, 0, stream>>>(segSrc[v], segIn, cstart[s], 6, 6);
      conv(segIn, 10 * 6 * 3, sgW[v][0], sgB[v][0], bufA, 1824, 10, 6, 3, 9);
      conv(bufA, 10 * 6 * 9, sgW[v][1], sgB[v][1], bufB, 1824, 10, 6, 9, 9);
      conv(bufB, 10 * 6 * 9, sgW[v][2], sgB[v][2], bufA, 1824, 10, 6, 9, 9);
      pool(bufA, nPool[s], 1824, 10, 6, 9);
    }
    seg_pack<<<gridFor(16L * 92340), 256, 0, stream>>>(segPool, nPool[0], nPool[1], nPool[2], featS[v]);
  }

  // ---- 4) Integ dense stacks -> concat (16, 2400) ----
  const float* featIn[8] = { featO[0], featO[1], featO[2], featO[3], featO[4],
                             featS[0], featS[1], featS[2] };
  const int dIn[8] = { 60750, 60750, 37908, 37908, 37908, 92340, 92340, 92340 };
  for (int i = 0; i < 8; ++i) {
    gemm(featIn[i], dIn[i], integW[i][0], integB[i][0], t300a, 300, dIn[i], 300);
    gemm(t300a, 300, integW[i][1], integB[i][1], t300b, 300, 300, 300);
    gemm(t300b, 300, integW[i][2], integB[i][2], t300a, 300, 300, 300);
    gemm(t300a, 300, integW[i][3], integB[i][3], concatBuf + i * 300, 2400, 300, 300);
  }

  // ---- 5) Ending stack -> d_out (16, 4) ----
  gemm(concatBuf, 2400, endW[0], endB[0], e200, 200, 2400, 200);
  gemm(e200, 200, endW[1], endB[1], e100, 100, 200, 100);
  gemm(e100, 100, endW[2], endB[2], e50, 50, 100, 50);
  gemm(e50, 50, endW[3], endB[3], e20, 20, 50, 20);
  gemm(e20, 20, endW[4], endB[4], (float*)d_out, 4, 20, 4);
}
